// ExplainerNetwork_85650237817507
// MI455X (gfx1250) — matrix operand order repaired and checked by reading, not compiled, not run on hardware
//
#include <hip/hip_runtime.h>

// ---------------------------------------------------------------------------
// ExplainerNetwork forward on MI455X (gfx1250, wave32, WMMA).
// Edge MLP (3.2M x [88->32->32]) is the hot loop: each wave processes 8
// consecutive 16-edge tiles so the f16 B fragments / biases stay in registers
// across tiles. Node table + n_up accumulator are L2-resident (192 MB L2).
// ---------------------------------------------------------------------------

typedef _Float16 v8h  __attribute__((ext_vector_type(8)));
typedef _Float16 v16h __attribute__((ext_vector_type(16)));
typedef float    v8f  __attribute__((ext_vector_type(8)));

#define N_NODES_C 100000
#define N_EDGES_C 3200000
#define NODE_F_C  39
#define EDGE_F_C  10
#define HID_C     32
#define KPAD_C    96          // padded K for 88-wide (edge) and 71-wide (node) GEMM1

#define EDGE_TPW  8           // tiles per wave, edge kernel (200000 / (4*8) = 6250 blocks)
#define NODE_TPW  5           // tiles per wave, node kernel (6250 / (2*5) = 625 blocks)

// workspace layout (bytes)
#define WS_BFRAG_E1 0                 // 6 frags * 32 lanes * 16 halfs = 6144 B
#define WS_BFRAG_E2 6144              // 2 frags                       = 2048 B
#define WS_BFRAG_N1 8192              // 6 frags                       = 6144 B
#define WS_NH       14336             // N_NODES * 40 halfs = 8,000,000 B (rows 16B aligned)
#define WS_NUP      8014336           // N_NODES * 32 f32  = 12,800,000 B

// Padded-K row mapping for the edge GEMM1 input: concat(n[src](39), e(10), n[dst](39))
//   K  0..38 : n[src]       K 39    : 0
//   K 40..49 : e            K 50..55: 0
//   K 56..94 : n[dst]       K 95    : 0
__device__ __forceinline__ int edge_k_row(int k) {
    if (k < 39) return k;
    if (k < 40) return -1;
    if (k < 50) return 39 + (k - 40);
    if (k < 56) return -1;
    if (k < 95) return 49 + (k - 56);
    return -1;
}

__device__ __forceinline__ float fast_tanh(float x) {
    // tanh(x) = 1 - 2/(exp(2x)+1); saturates correctly at +-1 for |x| large.
    float e2 = __expf(2.0f * x);
    return 1.0f - 2.0f * __builtin_amdgcn_rcpf(e2 + 1.0f);
}

// ---------------------------------------------------------------------------
// Kernel 0a: pack f32 weights into per-lane f16 B-fragment layout.
// B fragment (16-bit, 32x16 KxN): lane<16 -> col=lane,    K = 0..7,16..23
//                                 lane>=16 -> col=lane-16, K = 8..15,24..31
// Fragment f stored as bfrag[(f*32+lane)*16 + j], j = K within lane chunk.
// ---------------------------------------------------------------------------
__global__ __launch_bounds__(448)
void pack_weights_kernel(const float* __restrict__ ew1, const float* __restrict__ ew2,
                         const float* __restrict__ nw1, _Float16* __restrict__ ws) {
    const int t = threadIdx.x;
    _Float16* b1 = ws + WS_BFRAG_E1 / 2;
    _Float16* b2 = ws + WS_BFRAG_E2 / 2;
    _Float16* bn = ws + WS_BFRAG_N1 / 2;
    if (t < 192) {                          // ew1: 6 fragments (3 k-steps x 2 n-tiles)
        int f = t >> 5, lane = t & 31;
        int kk = f >> 1, nt = f & 1;
        int n = nt * 16 + (lane & 15);
        #pragma unroll
        for (int j = 0; j < 16; ++j) {
            int k = kk * 32 + ((lane < 16) ? 0 : 8) + (j < 8 ? j : j + 8);
            int r = edge_k_row(k);
            float v = (r < 0) ? 0.0f : ew1[r * HID_C + n];
            b1[(f * 32 + lane) * 16 + j] = (_Float16)v;
        }
    } else if (t < 256) {                   // ew2: 2 fragments (K=32)
        int tt = t - 192, f = tt >> 5, lane = tt & 31;
        int n = f * 16 + (lane & 15);
        #pragma unroll
        for (int j = 0; j < 16; ++j) {
            int k = ((lane < 16) ? 0 : 8) + (j < 8 ? j : j + 8);
            b2[(f * 32 + lane) * 16 + j] = (_Float16)ew2[k * HID_C + n];
        }
    } else if (t < 448) {                   // nw1: 6 fragments, K 0..70 valid
        int tt = t - 256, f = tt >> 5, lane = tt & 31;
        int kk = f >> 1, nt = f & 1;
        int n = nt * 16 + (lane & 15);
        #pragma unroll
        for (int j = 0; j < 16; ++j) {
            int k = kk * 32 + ((lane < 16) ? 0 : 8) + (j < 8 ? j : j + 8);
            float v = (k < 71) ? nw1[k * HID_C + n] : 0.0f;
            bn[(f * 32 + lane) * 16 + j] = (_Float16)v;
        }
    }
}

// Kernel 0b: node table f32 -> f16, rows padded 39 -> 40 (80 B, 16B-aligned rows)
__global__ void convert_nodes_kernel(const float* __restrict__ n, _Float16* __restrict__ nh) {
    int i = blockIdx.x * blockDim.x + threadIdx.x;
    if (i >= N_NODES_C * 40) return;
    int g = i / 40, f = i % 40;
    nh[i] = (f < NODE_F_C) ? (_Float16)n[g * NODE_F_C + f] : (_Float16)0.0f;
}

// Kernel 0c: zero the scatter accumulator (must not rely on stale workspace)
__global__ void zero_nup_kernel(float* __restrict__ nup) {
    int i = blockIdx.x * blockDim.x + threadIdx.x;
    if (i < N_NODES_C * HID_C) nup[i] = 0.0f;
}

// ---------------------------------------------------------------------------
// Edge kernel: 4 waves/block, EDGE_TPW consecutive 16-edge tiles per wave.
//   GEMM1: [16x96]f16 x [96x32]f16 -> [16x32]f32   (3 k-steps x 2 n-tiles WMMA)
//   tanh -> LDS -> GEMM2: [16x32] x [32x32]        (2 WMMA)
//   tanh -> atomic scatter-add into n_up[src]
// ---------------------------------------------------------------------------
__global__ __launch_bounds__(128)
void edge_mlp_kernel(const _Float16* __restrict__ nh, const float* __restrict__ e,
                     const int* __restrict__ ei,
                     const float* __restrict__ eb1, const float* __restrict__ eb2,
                     const _Float16* __restrict__ bfrag1, const _Float16* __restrict__ bfrag2,
                     float* __restrict__ nup) {
    __shared__ _Float16 stage[4][16][KPAD_C];   // 12288 B : edge_in rows (f16)
    __shared__ _Float16 htile[4][16][HID_C];    //  4096 B : hidden activations
    __shared__ int      sidx[4][16];            //   256 B : src index per row

    const int lane  = threadIdx.x & 31;
    const int wv    = threadIdx.x >> 5;
    const int row   = lane & 15;                // edge row within 16-edge tile
    const int hi    = lane >> 4;                // which half of the row this lane stages
    const int col   = lane & 15;                // C/D-layout column for this lane
    const int tile0 = (blockIdx.x * 4 + wv) * EDGE_TPW;

    // ---- per-wave invariants: zero padding, B fragments, biases ----
    {
        v8h z = {};
        #pragma unroll
        for (int c = 0; c < 6; ++c)             // each lane zeroes 48 halfs of its row
            *(v8h*)&stage[wv][row][hi * 48 + c * 8] = z;
    }
    v16h be1[6], be2[2];
    #pragma unroll
    for (int f = 0; f < 6; ++f) be1[f] = *(const v16h*)(bfrag1 + (f * 32 + lane) * 16);
    #pragma unroll
    for (int f = 0; f < 2; ++f) be2[f] = *(const v16h*)(bfrag2 + (f * 32 + lane) * 16);
    const float b1lo = eb1[col], b1hi = eb1[col + 16];
    const float b2lo = eb2[col], b2hi = eb2[col + 16];

    for (int t = 0; t < EDGE_TPW; ++t) {
        const int tile = tile0 + t;
        const int ge   = tile * 16 + row;

        // ---- stage gathered features (zero pad K50..55 persists from above) ----
        const int src = ei[ge];
        const int dst = ei[N_EDGES_C + ge];
        if (hi == 0) {
            sidx[wv][row] = src;
            const v8h* s = (const v8h*)(nh + src * 40);    // 16B-aligned row
            #pragma unroll
            for (int c = 0; c < 5; ++c) *(v8h*)&stage[wv][row][c * 8] = s[c];   // K 0..39
            const float* ep = e + ge * EDGE_F_C;
            #pragma unroll
            for (int c = 0; c < 10; ++c)                   // K 40..49
                stage[wv][row][40 + c] = (_Float16)ep[c];
            if (t + 1 < EDGE_TPW)                          // prefetch next tile's e row
                __builtin_prefetch(ep + 16 * EDGE_F_C, 0, 1);
        } else {
            const v8h* s = (const v8h*)(nh + dst * 40);
            #pragma unroll
            for (int c = 0; c < 5; ++c) *(v8h*)&stage[wv][row][56 + c * 8] = s[c]; // K 56..95
        }

        // ---- GEMM1: A fragment from LDS per 16-bit 16x32 layout table ----
        union { v16h v; struct { v8h lo, hi; } p; } a;
        v8f acc0 = {}, acc1 = {};
        #pragma unroll
        for (int kk = 0; kk < 3; ++kk) {
            const int kb = kk * 32 + ((lane < 16) ? 0 : 8);
            a.p.lo = *(const v8h*)&stage[wv][row][kb];        // K kb..kb+7
            a.p.hi = *(const v8h*)&stage[wv][row][kb + 16];   // K kb+16..kb+23
            acc0 = __builtin_amdgcn_wmma_f32_16x16x32_f16(false, a.v, false, be1[kk * 2 + 0],
                                                          (short)0, acc0, false, false);
            acc1 = __builtin_amdgcn_wmma_f32_16x16x32_f16(false, a.v, false, be1[kk * 2 + 1],
                                                          (short)0, acc1, false, false);
        }

        // ---- bias + tanh, write h tile to LDS (C-layout: lane holds col, 8 rows) ----
        #pragma unroll
        for (int r = 0; r < 8; ++r) {
            const int m = r + hi * 8;
            htile[wv][m][col]      = (_Float16)fast_tanh(acc0[r] + b1lo);
            htile[wv][m][col + 16] = (_Float16)fast_tanh(acc1[r] + b1hi);
        }

        // ---- GEMM2 (K=32): A fragment from h tile ----
        v8f o0 = {}, o1 = {};
        {
            const int kb = (lane < 16) ? 0 : 8;
            a.p.lo = *(const v8h*)&htile[wv][row][kb];
            a.p.hi = *(const v8h*)&htile[wv][row][kb + 16];
            o0 = __builtin_amdgcn_wmma_f32_16x16x32_f16(false, a.v, false, be2[0],
                                                        (short)0, o0, false, false);
            o1 = __builtin_amdgcn_wmma_f32_16x16x32_f16(false, a.v, false, be2[1],
                                                        (short)0, o1, false, false);
        }

        // ---- bias + tanh + scatter-add to n_up[src] (L2-resident f32 atomics) ----
        #pragma unroll
        for (int r = 0; r < 8; ++r) {
            const int m = r + hi * 8;
            const int s = sidx[wv][m];
            atomicAdd(&nup[s * HID_C + col],      fast_tanh(o0[r] + b2lo));
            atomicAdd(&nup[s * HID_C + col + 16], fast_tanh(o1[r] + b2hi));
        }
    }
}

// ---------------------------------------------------------------------------
// Node kernel: 2 waves/block, NODE_TPW tiles of 16 nodes per wave.
//   GEMM: concat(n_up(32), n(39)) padded to 96 -> 32, tanh, then 32->1 dot.
// ---------------------------------------------------------------------------
__global__ __launch_bounds__(64)
void node_mlp_kernel(const float* __restrict__ nup, const float* __restrict__ n,
                     const float* __restrict__ nb1, const float* __restrict__ nw2,
                     const float* __restrict__ nb2, const _Float16* __restrict__ bfragN,
                     float* __restrict__ out) {
    __shared__ _Float16 stage[2][16][KPAD_C];   // 6144 B
    __shared__ float    hn[2][16][HID_C];       // 4096 B

    const int lane  = threadIdx.x & 31;
    const int wv    = threadIdx.x >> 5;
    const int row   = lane & 15;
    const int hi    = lane >> 4;
    const int col   = lane & 15;
    const int tile0 = (blockIdx.x * 2 + wv) * NODE_TPW;

    // per-wave invariants
    {
        v8h z = {};
        #pragma unroll
        for (int c = 0; c < 6; ++c)
            *(v8h*)&stage[wv][row][hi * 48 + c * 8] = z;
    }
    v16h bn[6];
    #pragma unroll
    for (int f = 0; f < 6; ++f) bn[f] = *(const v16h*)(bfragN + (f * 32 + lane) * 16);
    const float b0 = nb1[col], b1 = nb1[col + 16];
    const float bias2 = nb2[0];
    float w2[HID_C];
    #pragma unroll
    for (int c = 0; c < HID_C; ++c) w2[c] = nw2[c];

    for (int t = 0; t < NODE_TPW; ++t) {
        const int tile = tile0 + t;
        const int g    = tile * 16 + row;

        if (hi == 0) {                           // K 0..31 : n_up (f32 -> f16)
            const float* s = nup + g * HID_C;
            #pragma unroll
            for (int c = 0; c < 32; ++c) stage[wv][row][c] = (_Float16)s[c];
        } else {                                 // K 32..70 : n ; K 71..95 stay zero
            const float* s = n + g * NODE_F_C;
            #pragma unroll
            for (int c = 0; c < 39; ++c) stage[wv][row][32 + c] = (_Float16)s[c];
        }

        union { v16h v; struct { v8h lo, hi; } p; } a;
        v8f acc0 = {}, acc1 = {};
        #pragma unroll
        for (int kk = 0; kk < 3; ++kk) {
            const int kb = kk * 32 + ((lane < 16) ? 0 : 8);
            a.p.lo = *(const v8h*)&stage[wv][row][kb];
            a.p.hi = *(const v8h*)&stage[wv][row][kb + 16];
            acc0 = __builtin_amdgcn_wmma_f32_16x16x32_f16(false, a.v, false, bn[kk * 2 + 0],
                                                          (short)0, acc0, false, false);
            acc1 = __builtin_amdgcn_wmma_f32_16x16x32_f16(false, a.v, false, bn[kk * 2 + 1],
                                                          (short)0, acc1, false, false);
        }

        #pragma unroll
        for (int r = 0; r < 8; ++r) {
            const int m = r + hi * 8;
            hn[wv][m][col]      = fast_tanh(acc0[r] + b0);
            hn[wv][m][col + 16] = fast_tanh(acc1[r] + b1);
        }

        // final 32 -> 1 layer: one row per lane 0..15
        if (lane < 16) {
            float accv = bias2;
            #pragma unroll
            for (int c = 0; c < HID_C; ++c) accv += hn[wv][lane][c] * w2[c];
            out[tile * 16 + lane] = accv;
        }
    }
}

// ---------------------------------------------------------------------------
extern "C" void kernel_launch(void* const* d_in, const int* in_sizes, int n_in,
                              void* d_out, int out_size, void* d_ws, size_t ws_size,
                              hipStream_t stream) {
    const float* n   = (const float*)d_in[0];
    const float* e   = (const float*)d_in[1];
    const int*   ei  = (const int*)d_in[2];
    // d_in[3] = batch (unused by reference math)
    const float* ew1 = (const float*)d_in[4];
    const float* eb1 = (const float*)d_in[5];
    const float* ew2 = (const float*)d_in[6];
    const float* eb2 = (const float*)d_in[7];
    const float* nw1 = (const float*)d_in[8];
    const float* nb1 = (const float*)d_in[9];
    const float* nw2 = (const float*)d_in[10];
    const float* nb2 = (const float*)d_in[11];

    char* ws = (char*)d_ws;
    _Float16* nh  = (_Float16*)(ws + WS_NH);
    float*    nup = (float*)(ws + WS_NUP);
    float*    out = (float*)d_out;

    pack_weights_kernel<<<1, 448, 0, stream>>>(ew1, ew2, nw1, (_Float16*)ws);
    convert_nodes_kernel<<<(N_NODES_C * 40 + 255) / 256, 256, 0, stream>>>(n, nh);
    zero_nup_kernel<<<(N_NODES_C * HID_C + 255) / 256, 256, 0, stream>>>(nup);

    // 200000 tiles / (4 waves * 8 tiles) = 6250 blocks (exact)
    edge_mlp_kernel<<<N_EDGES_C / 16 / (4 * EDGE_TPW), 128, 0, stream>>>(
        nh, e, ei, eb1, eb2,
        (const _Float16*)(ws + WS_BFRAG_E1),
        (const _Float16*)(ws + WS_BFRAG_E2), nup);

    // 6250 tiles / (2 waves * 5 tiles) = 625 blocks (exact)
    node_mlp_kernel<<<N_NODES_C / 16 / (2 * NODE_TPW), 64, 0, stream>>>(
        nup, n, nb1, nw2, nb2,
        (const _Float16*)(ws + WS_BFRAG_N1), out);
}
